// PredictorMamba_63505386438825
// MI455X (gfx1250) — matrix operand order repaired and checked by reading, not compiled, not run on hardware
//
#include <hip/hip_runtime.h>
#include <hip/hip_bf16.h>
#include <math.h>

// Problem dims (fixed by the reference)
#define B_    4
#define L_    1024
#define DM_   768
#define DI_   1536
#define DS_   16
#define DC_   4
#define DTR_  48
#define DTP_  64           // dt_rank padded to a K multiple of 32
#define MROWS (B_ * L_)    // 4096
#define MASK_RATIO_ 0.6f

typedef __attribute__((ext_vector_type(16))) __bf16 v16bf;
typedef __attribute__((ext_vector_type(8)))  __bf16 v8bf;
typedef __attribute__((ext_vector_type(8)))  float  v8f;

__device__ __forceinline__ float silu_f(float x)     { return x / (1.0f + __expf(-x)); }
__device__ __forceinline__ float softplus_f(float x) { return (x > 20.0f) ? x : log1pf(__expf(x)); }

__device__ __forceinline__ v16bf cat8(v8bf lo, v8bf hi) {
    return __builtin_shufflevector(lo, hi,
        0, 1, 2, 3, 4, 5, 6, 7, 8, 9, 10, 11, 12, 13, 14, 15);
}

// ---------------------------------------------------------------------------
// K0: mask-token substitution + fp32 -> bf16 for the in_proj GEMM A operand.
// Also writes the boolean mask output (second tuple element).
// ---------------------------------------------------------------------------
__global__ __launch_bounds__(256) void mask_embed_kernel(
    const float* __restrict__ x, const float* __restrict__ noise,
    const float* __restrict__ mask_token, __bf16* __restrict__ xmbf,
    unsigned char* __restrict__ mask_out)
{
    int idx = blockIdx.x * blockDim.x + threadIdx.x;
    if (idx >= MROWS * DM_) return;
    int m = idx / DM_;
    int d = idx - m * DM_;
    bool mk = noise[m] < MASK_RATIO_;
    float v = mk ? mask_token[d] : x[idx];
    xmbf[idx] = (__bf16)v;
    if (d == 0) mask_out[m] = mk ? 1 : 0;
}

// ---------------------------------------------------------------------------
// fp32 -> bf16 weight conversion (weights stay resident in L2 afterwards)
// ---------------------------------------------------------------------------
__global__ __launch_bounds__(256) void cvt_f32_bf16_kernel(
    const float* __restrict__ src, __bf16* __restrict__ dst, int n)
{
    int i = blockIdx.x * blockDim.x + threadIdx.x;
    if (i < n) dst[i] = (__bf16)src[i];
}

// dt_proj_w (DI_ x 48) -> bf16 padded to (DI_ x 64) with zero K-tail
__global__ __launch_bounds__(256) void pad_dtw_kernel(
    const float* __restrict__ w, __bf16* __restrict__ dst)
{
    int i = blockIdx.x * blockDim.x + threadIdx.x;
    if (i >= DI_ * DTP_) return;
    int r = i / DTP_, c = i - r * DTP_;
    dst[i] = (c < DTR_) ? (__bf16)w[r * DTR_ + c] : (__bf16)0.0f;
}

// dt slice of dbl (MROWS x 80, cols 0..47) -> bf16 padded (MROWS x 64)
__global__ __launch_bounds__(256) void extract_dt_kernel(
    const float* __restrict__ dbl, __bf16* __restrict__ dst)
{
    int i = blockIdx.x * blockDim.x + threadIdx.x;
    if (i >= MROWS * DTP_) return;
    int r = i / DTP_, c = i - r * DTP_;
    dst[i] = (c < DTR_) ? (__bf16)dbl[r * (DTR_ + 2 * DS_) + c] : (__bf16)0.0f;
}

// ---------------------------------------------------------------------------
// Generic bf16 WMMA GEMM:  C(MxN, f32) = A(MxK) * Bw(NxK)^T
// One wave computes a 32(M) x 64(N) tile: 2 A frags x 4 B frags -> 8 WMMAs
// per K=32 step.  Tile indices/subtile count are forced into SGPRs via
// readfirstlane so all guards are scalar branches (no EXEC writes), and
// ALL accumulator accesses use compile-time indices so acc stays in VGPRs
// (no scratch spills).
// mode 0: plain store.  mode 1: softplus(acc + bias[n])  (dt_proj epilogue).
// Requires: M % 32 == 0, K % 32 == 0, N % 16 == 0.
// ---------------------------------------------------------------------------
__global__ __launch_bounds__(256) void gemm_bf16_wmma(
    const __bf16* __restrict__ A, const __bf16* __restrict__ Bw,
    float* __restrict__ C, const float* __restrict__ bias, int mode,
    int M, int N, int K)
{
    const int wave = __builtin_amdgcn_readfirstlane(
        (int)((blockIdx.x * blockDim.x + threadIdx.x) >> 5));
    const int lane = threadIdx.x & 31;
    const int mtiles = M >> 5;                 // 32-row tiles
    const int nt64   = (N + 63) >> 6;
    if (wave >= mtiles * nt64) return;
    const int mt = wave / nt64;
    const int nt = wave - mt * nt64;

    // SGPR-resident count of valid 16-wide N subtiles in this 64-wide tile
    const int njsub = __builtin_amdgcn_readfirstlane(min(4, (N - nt * 64) >> 4));

    const int koff_a = (lane >= 16) ? 16 : 0;  // A: upper half-lanes hold K+16..
    const int koff_b = (lane >= 16) ? 16 : 0;  // B: upper half-lanes hold K+16..
    const __bf16* Arow0 = A + (size_t)(mt * 32 + (lane & 15)) * K + koff_a;
    const __bf16* Arow1 = Arow0 + (size_t)16 * K;
    const __bf16* Bcol  = Bw + (size_t)(nt * 64 + (lane & 15)) * K + koff_b;

    v8f acc[2][4];
#pragma unroll
    for (int i = 0; i < 2; ++i)
#pragma unroll
        for (int j = 0; j < 4; ++j)
#pragma unroll
            for (int r = 0; r < 8; ++r) acc[i][j][r] = 0.0f;

    if (njsub == 4) {
        // ---- hot path: full 32x64 tile, branch-free inner loop ----
        for (int k0 = 0; k0 < K; k0 += 32) {
            // issue ALL fragment loads first (clause-able), then all WMMAs
            v8bf a0lo = *(const v8bf*)(Arow0 + k0);
            v8bf a0hi = *(const v8bf*)(Arow0 + k0 + 8);
            v8bf a1lo = *(const v8bf*)(Arow1 + k0);
            v8bf a1hi = *(const v8bf*)(Arow1 + k0 + 8);
            v16bf b0 = *(const v16bf*)(Bcol + (size_t)0  * K + k0);
            v16bf b1 = *(const v16bf*)(Bcol + (size_t)16 * K + k0);
            v16bf b2 = *(const v16bf*)(Bcol + (size_t)32 * K + k0);
            v16bf b3 = *(const v16bf*)(Bcol + (size_t)48 * K + k0);
            v16bf af0 = cat8(a0lo, a0hi);
            v16bf af1 = cat8(a1lo, a1hi);

            acc[0][0] = __builtin_amdgcn_wmma_f32_16x16x32_bf16(
                false, af0, false, b0, (short)0, acc[0][0], false, false);
            acc[1][0] = __builtin_amdgcn_wmma_f32_16x16x32_bf16(
                false, af1, false, b0, (short)0, acc[1][0], false, false);
            acc[0][1] = __builtin_amdgcn_wmma_f32_16x16x32_bf16(
                false, af0, false, b1, (short)0, acc[0][1], false, false);
            acc[1][1] = __builtin_amdgcn_wmma_f32_16x16x32_bf16(
                false, af1, false, b1, (short)0, acc[1][1], false, false);
            acc[0][2] = __builtin_amdgcn_wmma_f32_16x16x32_bf16(
                false, af0, false, b2, (short)0, acc[0][2], false, false);
            acc[1][2] = __builtin_amdgcn_wmma_f32_16x16x32_bf16(
                false, af1, false, b2, (short)0, acc[1][2], false, false);
            acc[0][3] = __builtin_amdgcn_wmma_f32_16x16x32_bf16(
                false, af0, false, b3, (short)0, acc[0][3], false, false);
            acc[1][3] = __builtin_amdgcn_wmma_f32_16x16x32_bf16(
                false, af1, false, b3, (short)0, acc[1][3], false, false);
        }
    } else {
        // ---- tail path: ragged N tile (only the N=80 GEMM hits this) ----
        // NOTE: fully unrolled with constant indices; guard is a scalar
        // branch (njsub is SGPR), so acc stays register-promoted.
        for (int k0 = 0; k0 < K; k0 += 32) {
            v16bf af0 = cat8(*(const v8bf*)(Arow0 + k0),
                             *(const v8bf*)(Arow0 + k0 + 8));
            v16bf af1 = cat8(*(const v8bf*)(Arow1 + k0),
                             *(const v8bf*)(Arow1 + k0 + 8));
#pragma unroll
            for (int j = 0; j < 4; ++j) {
                if (j < njsub) {
                    v16bf bfrag = *(const v16bf*)(Bcol + (size_t)(j * 16) * K + k0);
                    acc[0][j] = __builtin_amdgcn_wmma_f32_16x16x32_bf16(
                        false, af0, false, bfrag, (short)0, acc[0][j], false, false);
                    acc[1][j] = __builtin_amdgcn_wmma_f32_16x16x32_bf16(
                        false, af1, false, bfrag, (short)0, acc[1][j], false, false);
                }
            }
        }
    }

    // C/D layout: VGPR r -> row (lane<16 ? r : r+8), col = lane&15
    // Fully unrolled, constant acc indices; scalar guard per subtile.
    const int rbase = mt * 32 + ((lane < 16) ? 0 : 8);
#pragma unroll
    for (int i = 0; i < 2; ++i) {
#pragma unroll
        for (int j = 0; j < 4; ++j) {
            if (j < njsub) {
                const int col = nt * 64 + j * 16 + (lane & 15);
                float* Cp = C + (size_t)(rbase + i * 16) * N + col;
                if (mode == 1) {
                    const float bb = bias[col];
#pragma unroll
                    for (int r = 0; r < 8; ++r)
                        Cp[(size_t)r * N] = softplus_f(acc[i][j][r] + bb);
                } else {
#pragma unroll
                    for (int r = 0; r < 8; ++r)
                        Cp[(size_t)r * N] = acc[i][j][r];
                }
            }
        }
    }
}

// ---------------------------------------------------------------------------
// K2: causal depthwise conv (D_CONV=4, left pad 3) + bias + SiLU.
// Reads xi half of xz (MROWS x 3072), writes xi f32 (for scan) and bf16
// (A operand of the x_proj GEMM).
// ---------------------------------------------------------------------------
__global__ __launch_bounds__(256) void conv_silu_kernel(
    const float* __restrict__ xz, const float* __restrict__ conv_w,
    const float* __restrict__ conv_b, float* __restrict__ xi_f32,
    __bf16* __restrict__ xi_bf)
{
    int idx = blockIdx.x * blockDim.x + threadIdx.x;
    if (idx >= MROWS * DI_) return;
    int row = idx / DI_;           // b*L + l
    int c   = idx - row * DI_;
    int b   = row / L_;
    int l   = row - b * L_;

    float acc = conv_b[c];
#pragma unroll
    for (int j = 0; j < DC_; ++j) {
        int ls = l - (DC_ - 1) + j;
        if (ls >= 0)
            acc += conv_w[c * DC_ + j] * xz[((size_t)(b * L_ + ls)) * (2 * DI_) + c];
    }
    float v = silu_f(acc);
    xi_f32[idx] = v;
    xi_bf[idx]  = (__bf16)v;
}

// ---------------------------------------------------------------------------
// K5: fused selective scan.  One thread per (b, channel): 16-state recurrence
// in registers, 1024 sequential steps.  Epilogue fuses  + u*D_skip, *silu(z)
// and the bf16 downcast for the out_proj GEMM A operand.
// ---------------------------------------------------------------------------
__global__ __launch_bounds__(256) void scan_kernel(
    const float* __restrict__ xi, const float* __restrict__ delta,
    const float* __restrict__ dbl, const float* __restrict__ xz,
    const float* __restrict__ A_log, const float* __restrict__ D_skip,
    __bf16* __restrict__ ybf)
{
    int tid = blockIdx.x * blockDim.x + threadIdx.x;
    if (tid >= B_ * DI_) return;
    int b = tid / DI_;
    int d = tid - b * DI_;

    float Arow[DS_];
#pragma unroll
    for (int s = 0; s < DS_; ++s) Arow[s] = -__expf(A_log[d * DS_ + s]);
    const float dsk = D_skip[d];

    float h[DS_];
#pragma unroll
    for (int s = 0; s < DS_; ++s) h[s] = 0.0f;

    for (int t = 0; t < L_; ++t) {
        const size_t row = (size_t)b * L_ + t;
        const float u   = xi[row * DI_ + d];
        const float dlt = delta[row * DI_ + d];
        const float du  = dlt * u;
        const float* Bs = dbl + row * (DTR_ + 2 * DS_) + DTR_;
        const float* Cs = Bs + DS_;
        float y = 0.0f;
#pragma unroll
        for (int s = 0; s < DS_; ++s) {
            const float dA = __expf(dlt * Arow[s]);
            h[s] = dA * h[s] + du * Bs[s];
            y += h[s] * Cs[s];
        }
        y += u * dsk;
        const float z = xz[row * (2 * DI_) + DI_ + d];
        y *= silu_f(z);
        ybf[row * DI_ + d] = (__bf16)y;
    }
}

// ---------------------------------------------------------------------------
// K7: LayerNorm over DM=768 per row.  One 256-thread block per row.
// ---------------------------------------------------------------------------
__global__ __launch_bounds__(256) void layernorm_kernel(
    const float* __restrict__ src, const float* __restrict__ g,
    const float* __restrict__ bb, float* __restrict__ out)
{
    __shared__ float ssum[256];
    __shared__ float ssq[256];
    const int row = blockIdx.x;
    const int t   = threadIdx.x;
    const float* sr = src + (size_t)row * DM_;

    float v0 = sr[t], v1 = sr[t + 256], v2 = sr[t + 512];
    ssum[t] = v0 + v1 + v2;
    ssq[t]  = v0 * v0 + v1 * v1 + v2 * v2;
    __syncthreads();
    for (int off = 128; off > 0; off >>= 1) {
        if (t < off) { ssum[t] += ssum[t + off]; ssq[t] += ssq[t + off]; }
        __syncthreads();
    }
    const float mu  = ssum[0] * (1.0f / DM_);
    const float var = ssq[0] * (1.0f / DM_) - mu * mu;
    const float inv = rsqrtf(var + 1e-5f);

    float* orow = out + (size_t)row * DM_;
    orow[t]       = (v0 - mu) * inv * g[t]       + bb[t];
    orow[t + 256] = (v1 - mu) * inv * g[t + 256] + bb[t + 256];
    orow[t + 512] = (v2 - mu) * inv * g[t + 512] + bb[t + 512];
}

// ---------------------------------------------------------------------------
extern "C" void kernel_launch(void* const* d_in, const int* in_sizes, int n_in,
                              void* d_out, int out_size, void* d_ws, size_t ws_size,
                              hipStream_t stream)
{
    const float* x          = (const float*)d_in[0];
    const float* noise      = (const float*)d_in[1];
    const float* mask_token = (const float*)d_in[2];
    const float* in_proj_w  = (const float*)d_in[3];
    const float* conv_w     = (const float*)d_in[4];
    const float* conv_b     = (const float*)d_in[5];
    const float* x_proj_w   = (const float*)d_in[6];
    const float* dt_proj_w  = (const float*)d_in[7];
    const float* dt_proj_b  = (const float*)d_in[8];
    const float* A_log      = (const float*)d_in[9];
    const float* D_skip     = (const float*)d_in[10];
    const float* out_proj_w = (const float*)d_in[11];
    const float* ln_g       = (const float*)d_in[12];
    const float* ln_b       = (const float*)d_in[13];

    // Workspace carve-up (256B aligned). Total ~154 MB — fits the 192 MB L2.
    char* p = (char*)d_ws;
    auto alloc = [&](size_t bytes) -> char* {
        char* r = p; p += (bytes + 255) & ~(size_t)255; return r;
    };
    __bf16* xmbf   = (__bf16*)alloc((size_t)MROWS * DM_ * 2);          // masked input, bf16
    __bf16* w1bf   = (__bf16*)alloc((size_t)2 * DI_ * DM_ * 2);        // in_proj_w bf16
    float*  xz     = (float*) alloc((size_t)MROWS * 2 * DI_ * 4);      // xi | z
    float*  xif    = (float*) alloc((size_t)MROWS * DI_ * 4);          // conv+silu f32
    __bf16* xibf   = (__bf16*)alloc((size_t)MROWS * DI_ * 2);
    __bf16* xpwbf  = (__bf16*)alloc((size_t)(DTR_ + 2 * DS_) * DI_ * 2);
    float*  dbl    = (float*) alloc((size_t)MROWS * (DTR_ + 2 * DS_) * 4);
    __bf16* dtbf   = (__bf16*)alloc((size_t)MROWS * DTP_ * 2);         // dt, K-padded
    __bf16* dtwbf  = (__bf16*)alloc((size_t)DI_ * DTP_ * 2);
    float*  delta  = (float*) alloc((size_t)MROWS * DI_ * 4);
    __bf16* ybf    = (__bf16*)alloc((size_t)MROWS * DI_ * 2);          // gated scan output
    __bf16* owbf   = (__bf16*)alloc((size_t)DM_ * DI_ * 2);
    float*  outpre = (float*) alloc((size_t)MROWS * DM_ * 4);

    float*         out_f  = (float*)d_out;
    unsigned char* mask_o = (unsigned char*)d_out + (size_t)MROWS * DM_ * sizeof(float);

    auto gemm = [&](const __bf16* A, const __bf16* Bw, float* C,
                    const float* bias, int mode, int M, int N, int K) {
        int waves  = (M >> 5) * ((N + 63) >> 6);
        int blocks = (waves + 7) / 8;   // 8 waves per 256-thread block
        gemm_bf16_wmma<<<blocks, 256, 0, stream>>>(A, Bw, C, bias, mode, M, N, K);
    };

    // 0) mask + embed + bf16, and emit the bool mask output
    mask_embed_kernel<<<(MROWS * DM_ + 255) / 256, 256, 0, stream>>>(
        x, noise, mask_token, xmbf, mask_o);

    // weight conversions (cheap; keeps GEMM operands bf16-resident in L2)
    cvt_f32_bf16_kernel<<<(2 * DI_ * DM_ + 255) / 256, 256, 0, stream>>>(
        in_proj_w, w1bf, 2 * DI_ * DM_);
    cvt_f32_bf16_kernel<<<((DTR_ + 2 * DS_) * DI_ + 255) / 256, 256, 0, stream>>>(
        x_proj_w, xpwbf, (DTR_ + 2 * DS_) * DI_);
    cvt_f32_bf16_kernel<<<(DM_ * DI_ + 255) / 256, 256, 0, stream>>>(
        out_proj_w, owbf, DM_ * DI_);
    pad_dtw_kernel<<<(DI_ * DTP_ + 255) / 256, 256, 0, stream>>>(dt_proj_w, dtwbf);

    // 1) xz = xm @ in_proj_w^T            (4096 x 3072, K=768)
    gemm(xmbf, w1bf, xz, nullptr, 0, MROWS, 2 * DI_, DM_);

    // 2) causal depthwise conv + bias + SiLU
    conv_silu_kernel<<<(MROWS * DI_ + 255) / 256, 256, 0, stream>>>(
        xz, conv_w, conv_b, xif, xibf);

    // 3) dbl = xi @ x_proj_w^T            (4096 x 80, K=1536)
    gemm(xibf, xpwbf, dbl, nullptr, 0, MROWS, DTR_ + 2 * DS_, DI_);

    // 4) delta = softplus(dt @ dt_proj_w^T + b)   (4096 x 1536, K=64 padded)
    extract_dt_kernel<<<(MROWS * DTP_ + 255) / 256, 256, 0, stream>>>(dbl, dtbf);
    gemm(dtbf, dtwbf, delta, dt_proj_b, 1, MROWS, DI_, DTP_);

    // 5) fused selective scan + D-skip + SiLU(z) gate -> y (bf16)
    scan_kernel<<<(B_ * DI_ + 255) / 256, 256, 0, stream>>>(
        xif, delta, dbl, xz, A_log, D_skip, ybf);

    // 6) out_pre = y @ out_proj_w^T       (4096 x 768, K=1536)
    gemm(ybf, owbf, outpre, nullptr, 0, MROWS, DM_, DI_);

    // 7) LayerNorm -> d_out
    layernorm_kernel<<<MROWS, 256, 0, stream>>>(outpre, ln_g, ln_b, out_f);
}
